// coRNN_19155554140902
// MI455X (gfx1250) — matrix operand order repaired and checked by reading, not compiled, not run on hardware
//
#include <hip/hip_runtime.h>
#include <hip/hip_bf16.h>
#include <math.h>

// Problem constants (must match reference)
#define T_STEPS 1024
#define BATCH   128
#define N_INP   128
#define N_HID   512
#define DT_C    0.042f
#define GAMMA_C 2.7f
#define EPS_C   4.7f

// Derived tiling
#define KTOT    1152              // 512 (hz) + 512 (hy) + 128 (x)
#define KC      36                // KTOT / 32 k-chunks for wmma 16x16x32
#define NT      32                // N_HID / 16 n-tiles
#define WAVES   16                // 16 waves * wave32 = 512 threads
#define THREADS (WAVES * 32)
#define BTILE   16                // batch rows per workgroup (== WMMA M)
#define NGROUPS (BATCH / BTILE)   // 8 persistent workgroups
#define ROWPAD  1160              // 1152 + 8 bf16 pad -> row stride 580 dwords (580%64=4, no bank conflict)

typedef __attribute__((ext_vector_type(16))) __bf16 v16bf;
typedef __attribute__((ext_vector_type(8)))  float  v8f;

union AFrag {
    v16bf v;
    unsigned int u[8];
};
union BFrag {
    v16bf v;
    __bf16 e[16];
};

// ---------------------------------------------------------------------------
// Pack [Wh | Wi] (fp32, row-major over n) into bf16 WMMA B-fragment order:
//   frag(nt, kc) lane l holds column n = nt*16 + (l&15),
//   K values kc*32 + 16*(l>>4) + i  for i = 0..15 (contiguous pairs/VGPR).
// Flat layout: wp[(((nt*KC + kc)*32) + lane)*16 + i]
// Combined row: W[n][k] = Wh[n][k] for k<1024 (Wz then Wy), Wi[n][k-1024] else.
// ---------------------------------------------------------------------------
__global__ void pack_weights_kernel(const float* __restrict__ Wi,
                                    const float* __restrict__ Wh,
                                    __bf16* __restrict__ wp) {
    int tid = blockIdx.x * blockDim.x + threadIdx.x;
    if (tid >= NT * KC * 32) return;
    int lane = tid & 31;
    int kc   = (tid >> 5) % KC;
    int nt   = tid / (32 * KC);
    int r    = lane & 15;
    int hf   = lane >> 4;
    int n    = nt * 16 + r;

    BFrag f;
#pragma unroll
    for (int i = 0; i < 16; ++i) {
        int k = kc * 32 + hf * 16 + i;
        float v = (k < 1024) ? Wh[(size_t)n * 1024 + k]
                             : Wi[(size_t)n * N_INP + (k - 1024)];
        f.e[i] = (__bf16)v;
    }
    v16bf* dst = (v16bf*)(wp + (((size_t)(nt * KC + kc) * 32) + lane) * 16);
    *dst = f.v;
}

// ---------------------------------------------------------------------------
// Persistent scan kernel: 8 workgroups, one per 16-row batch tile.
// S (LDS, bf16): per batch row m, combined A-source row [hz(512)|hy(512)|x(128)].
// fp32 master state (hy,hz) stays in registers on the lane that owns the
// corresponding C/D fragment element: (m = j + 8*hf, n = ntile*16 + r).
// ---------------------------------------------------------------------------
__global__ __launch_bounds__(THREADS)
void corn_scan_kernel(const float* __restrict__ x,
                      const float* __restrict__ bi,
                      const __bf16* __restrict__ wp,
                      float* __restrict__ out) {
    __shared__ __bf16 S[BTILE][ROWPAD];

    const int tid  = threadIdx.x;
    const int lane = tid & 31;
    const int wave = tid >> 5;
    const int r    = lane & 15;
    const int hf   = lane >> 4;
    const int b0   = blockIdx.x * BTILE;
    const int ntg0 = wave * 2;          // each wave owns n-tiles ntg0, ntg0+1

    // ---- init: zero hz/hy region of S (cols [0,1024)) ----
    for (int idx = tid; idx < BTILE * 1024; idx += THREADS) {
        S[idx >> 10][idx & 1023] = (__bf16)0.0f;
    }
    // ---- stage x(t=0) into S cols [1024,1152) ----
    {
        int row = tid >> 5;             // 0..15
        int c0  = (tid & 31) * 4;       // 0..124
        float4 xv = *(const float4*)&x[((size_t)0 * BATCH + b0 + row) * N_INP + c0];
        S[row][1024 + c0 + 0] = (__bf16)xv.x;
        S[row][1024 + c0 + 1] = (__bf16)xv.y;
        S[row][1024 + c0 + 2] = (__bf16)xv.z;
        S[row][1024 + c0 + 3] = (__bf16)xv.w;
    }

    // fp32 register state, one (m,n) element per acc slot
    float hzr[2][8], hyr[2][8];
#pragma unroll
    for (int tI = 0; tI < 2; ++tI)
#pragma unroll
        for (int j = 0; j < 8; ++j) { hzr[tI][j] = 0.0f; hyr[tI][j] = 0.0f; }

    float bival[2];
    bival[0] = bi[(ntg0 + 0) * 16 + r];
    bival[1] = bi[(ntg0 + 1) * 16 + r];

    // per-lane base pointers into packed weight stream (L2-resident)
    const __bf16* wl0 = wp + (((size_t)((ntg0 + 0) * KC) * 32) + lane) * 16;
    const __bf16* wl1 = wp + (((size_t)((ntg0 + 1) * KC) * 32) + lane) * 16;

    __syncthreads();

    for (int t = 0; t < T_STEPS; ++t) {
        // ---- GEMM phase: pre = [hz|hy|x_t] @ W^T + bi ----
        v8f acc0, acc1;
#pragma unroll
        for (int j = 0; j < 8; ++j) { acc0[j] = bival[0]; acc1[j] = bival[1]; }

        for (int kc = 0; kc < KC; ++kc) {
            // Gather A fragment from LDS, contiguous K order (matches packed B):
            //   hf=0: K pairs {0,2,..,14}; hf=1: {16,18,..,30}
            AFrag a;
#pragma unroll
            for (int j = 0; j < 8; ++j) {
                int k2 = kc * 32 + 16 * hf + 2 * j;
                a.u[j] = *(const unsigned int*)&S[r][k2];
            }
            // Stream pre-packed B fragments (32B contiguous per lane)
            v16bf bv0 = *(const v16bf*)(wl0 + (size_t)kc * 32 * 16);
            v16bf bv1 = *(const v16bf*)(wl1 + (size_t)kc * 32 * 16);

            acc0 = __builtin_amdgcn_wmma_f32_16x16x32_bf16(
                false, a.v, false, bv0, (short)0, acc0, false, false);
            acc1 = __builtin_amdgcn_wmma_f32_16x16x32_bf16(
                false, a.v, false, bv1, (short)0, acc1, false, false);
        }

        __syncthreads();   // all reads of S done before state rewrite

        // ---- state update (leapfrog) on owned elements ----
#pragma unroll
        for (int tI = 0; tI < 2; ++tI) {
            int n = (ntg0 + tI) * 16 + r;
#pragma unroll
            for (int j = 0; j < 8; ++j) {
                int m = j + 8 * hf;
                float pre = (tI == 0) ? acc0[j] : acc1[j];
                float th  = tanhf(pre);
                float z = hzr[tI][j];
                float y = hyr[tI][j];
                z += DT_C * (th - GAMMA_C * y - EPS_C * z);
                y += DT_C * z;
                hzr[tI][j] = z;
                hyr[tI][j] = y;
                S[m][n]       = (__bf16)z;   // hz shadow for next step's A
                S[m][512 + n] = (__bf16)y;   // hy shadow
            }
        }

        // ---- stage x(t+1) while we're between barriers ----
        if (t + 1 < T_STEPS) {
            int row = tid >> 5;
            int c0  = (tid & 31) * 4;
            float4 xv = *(const float4*)&x[((size_t)(t + 1) * BATCH + b0 + row) * N_INP + c0];
            S[row][1024 + c0 + 0] = (__bf16)xv.x;
            S[row][1024 + c0 + 1] = (__bf16)xv.y;
            S[row][1024 + c0 + 2] = (__bf16)xv.z;
            S[row][1024 + c0 + 3] = (__bf16)xv.w;
        }

        __syncthreads();   // new state visible before next step's gathers
    }

    // ---- write final hy (B x N_HID, fp32) ----
#pragma unroll
    for (int tI = 0; tI < 2; ++tI) {
        int n = (ntg0 + tI) * 16 + r;
#pragma unroll
        for (int j = 0; j < 8; ++j) {
            int m = j + 8 * hf;
            out[(size_t)(b0 + m) * N_HID + n] = hyr[tI][j];
        }
    }
}

// ---------------------------------------------------------------------------
extern "C" void kernel_launch(void* const* d_in, const int* in_sizes, int n_in,
                              void* d_out, int out_size, void* d_ws, size_t ws_size,
                              hipStream_t stream) {
    const float* x  = (const float*)d_in[0];   // (T, B, N_INP) fp32
    const float* Wi = (const float*)d_in[1];   // (N_HID, N_INP) fp32
    const float* bi = (const float*)d_in[2];   // (N_HID,) fp32
    const float* Wh = (const float*)d_in[3];   // (N_HID, 2*N_HID) fp32

    __bf16* wp = (__bf16*)d_ws;                // packed bf16 weights, ~1.13 MB

    int packThreads = NT * KC * 32;            // 36864
    pack_weights_kernel<<<(packThreads + 255) / 256, 256, 0, stream>>>(Wi, Wh, wp);

    corn_scan_kernel<<<NGROUPS, THREADS, 0, stream>>>(x, bi, wp, (float*)d_out);
}
